// Model_32667521254286
// MI455X (gfx1250) — matrix operand order repaired and checked by reading, not compiled, not run on hardware
//
#include <hip/hip_runtime.h>
#include <hip/hip_bf16.h>
#include <math.h>

#define NN    50000
#define EE    800000
#define DD    256
#define GG    128
#define KTILES 8          // 256 / 32
#define NTILES 16         // 256 / 16
#define MTILES (NN / 16)  // 3125

typedef __bf16 bf16_t;
typedef __attribute__((ext_vector_type(16))) __bf16 bf16x16;
typedef __attribute__((ext_vector_type(8)))  float  v8f;

// ---------------------------------------------------------------------------
// Pack W (256x256 f32, row-major [k][n]) into per-lane WMMA B fragments,
// split into bf16 hi + lo parts.  Layout: [kb][nt][lane][e] (16 halves/lane).
// B 32x16 bf16 fragment: lanes 0-15 hold K=kb*32+{0..7,16..23}, lanes 16-31 hold
// K=kb*32+{8..15,24..31}; column N = nt*16 + (lane & 15).
// ---------------------------------------------------------------------------
__global__ void pack_w_kernel(const float* __restrict__ W,
                              bf16_t* __restrict__ Bh,
                              bf16_t* __restrict__ Bl) {
    int idx  = blockIdx.x * blockDim.x + threadIdx.x;      // 0 .. 65535
    int e    = idx & 15;
    int lane = (idx >> 4) & 31;
    int nt   = (idx >> 9) & 15;
    int kb   = idx >> 13;
    int k    = kb * 32 + ((lane < 16) ? 0 : 8) + ((e < 8) ? e : e + 8);
    int n    = nt * 16 + (lane & 15);
    float w  = W[k * DD + n];
    bf16_t hi = (bf16_t)w;
    Bh[idx] = hi;
    Bl[idx] = (bf16_t)(w - (float)hi);
}

// ---------------------------------------------------------------------------
// C = A @ W via bf16x3-split WMMA.  One wave per 16-row strip, 16 col tiles.
// A fragment (16x32 bf16): lane L<16 -> row L, K = {0..7, 16..23}+kbase;
// lane L>=16 -> row L-16, K = {8..15, 24..31}+kbase.
// ---------------------------------------------------------------------------
__global__ void __launch_bounds__(128)
gemm_bf16x3_kernel(const float* __restrict__ A,
                   const bf16_t* __restrict__ Bh,
                   const bf16_t* __restrict__ Bl,
                   float* __restrict__ C,
                   int ntiles_m) {
    const int wave = blockIdx.x * (blockDim.x >> 5) + (threadIdx.x >> 5);
    if (wave >= ntiles_m) return;                 // wave-uniform: EXEC stays full
    const int lane = threadIdx.x & 31;
    const int row0 = wave * 16;
    const int mrow = lane & 15;
    const int koff = (lane < 16) ? 0 : 8;
    const float* arow = A + (size_t)(row0 + mrow) * DD;

    const v8f vzero = {0.f, 0.f, 0.f, 0.f, 0.f, 0.f, 0.f, 0.f};
    v8f acc[NTILES];
#pragma unroll
    for (int t = 0; t < NTILES; ++t) acc[t] = vzero;

    for (int kb = 0; kb < KTILES; ++kb) {
        const int kbase = kb * 32;
        float af[16];
        const float* p0 = arow + kbase + koff;
        const float* p1 = arow + kbase + 16 + koff;
#pragma unroll
        for (int e = 0; e < 8; ++e) { af[e] = p0[e]; af[8 + e] = p1[e]; }

        bf16x16 a_hi, a_lo;
#pragma unroll
        for (int e = 0; e < 16; ++e) {
            bf16_t h = (bf16_t)af[e];
            a_hi[e] = h;
            a_lo[e] = (bf16_t)(af[e] - (float)h);
        }

        const size_t bbase = (size_t)kb * (NTILES * 32 * 16);
#pragma unroll
        for (int nt = 0; nt < NTILES; ++nt) {
            const size_t off = bbase + ((size_t)nt * 32 + lane) * 16;
            bf16x16 b_hi = *(const bf16x16*)(Bh + off);
            bf16x16 b_lo = *(const bf16x16*)(Bl + off);
            acc[nt] = __builtin_amdgcn_wmma_f32_16x16x32_bf16(
                false, a_hi, false, b_hi, (short)0, acc[nt], false, false);
            acc[nt] = __builtin_amdgcn_wmma_f32_16x16x32_bf16(
                false, a_hi, false, b_lo, (short)0, acc[nt], false, false);
            acc[nt] = __builtin_amdgcn_wmma_f32_16x16x32_bf16(
                false, a_lo, false, b_hi, (short)0, acc[nt], false, false);
        }
    }

    // C/D layout: VGPR v, lanes 0-15 -> M=v, N=lane; lanes 16-31 -> M=8+v.
    const int mbase = row0 + ((lane < 16) ? 0 : 8);
    const int ncol  = lane & 15;
#pragma unroll
    for (int nt = 0; nt < NTILES; ++nt) {
#pragma unroll
        for (int v = 0; v < 8; ++v) {
            C[(size_t)(mbase + v) * DD + nt * 16 + ncol] = acc[nt][v];
        }
    }
}

// ---------------------------------------------------------------------------
// Degree / normalization
// ---------------------------------------------------------------------------
__global__ void deg_init_kernel(float* __restrict__ deg) {
    int i = blockIdx.x * blockDim.x + threadIdx.x;
    if (i < NN) deg[i] = 1.0f;                    // self-loop contribution
}
__global__ void deg_accum_kernel(const int* __restrict__ ei, float* __restrict__ deg) {
    int e = blockIdx.x * blockDim.x + threadIdx.x;
    if (e < EE) atomicAdd(&deg[ei[EE + e]], 1.0f);
}
__global__ void dinv_kernel(float* __restrict__ deg) {
    int i = blockIdx.x * blockDim.x + threadIdx.x;
    if (i < NN) deg[i] = rsqrtf(deg[i]);          // deg >= 1 always
}

// ---------------------------------------------------------------------------
// Edge scatter: agg[dst] += hw[src] * dinv[src]*dinv[dst].  One block / edge.
// Blocks [EE, EE+NN) are the self-loops.
// ---------------------------------------------------------------------------
__global__ void __launch_bounds__(256)
scatter_kernel(const float* __restrict__ hw, const int* __restrict__ ei,
               const float* __restrict__ dinv, float* __restrict__ agg) {
    int e = blockIdx.x;
    int t = threadIdx.x;
    int s, d;
    if (e < EE) { s = ei[e]; d = ei[EE + e]; }
    else        { s = d = e - EE; }
    float nrm = dinv[s] * dinv[d];
    atomicAdd(&agg[(size_t)d * DD + t], hw[(size_t)s * DD + t] * nrm);
}

// ---------------------------------------------------------------------------
// BN pass 1: h_pre = prev + agg + bias (in place on agg), per-column partial
// sums / sums-of-squares accumulated with float atomics.
// ---------------------------------------------------------------------------
#define ROWS_PER_BLK 128
__global__ void __launch_bounds__(256)
bn_stats_kernel(float* __restrict__ hpre, const float* __restrict__ prev,
                const float* __restrict__ bias,
                float* __restrict__ sums, float* __restrict__ sumsq) {
    int d  = threadIdx.x;
    int r0 = blockIdx.x * ROWS_PER_BLK;
    float b = bias[d];
    float s = 0.f, sq = 0.f;
    for (int i = 0; i < ROWS_PER_BLK; ++i) {
        int r = r0 + i;
        if (r >= NN) break;
        size_t idx = (size_t)r * DD + d;
        float v = hpre[idx] + b + (prev ? prev[idx] : 0.f);
        hpre[idx] = v;
        s += v; sq += v * v;
    }
    atomicAdd(&sums[d], s);
    atomicAdd(&sumsq[d], sq);
}

// BN pass 2 + ELU
__global__ void __launch_bounds__(256)
bn_apply_kernel(const float* __restrict__ hpre,
                const float* __restrict__ sums, const float* __restrict__ sumsq,
                const float* __restrict__ gamma, const float* __restrict__ beta,
                float* __restrict__ hout) {
    int d  = threadIdx.x;
    float mu  = sums[d]  * (1.0f / NN);
    float var = sumsq[d] * (1.0f / NN) - mu * mu;
    float rs  = rsqrtf(var + 1e-5f);
    float g = gamma[d], be = beta[d];
    int r0 = blockIdx.x * ROWS_PER_BLK;
    for (int i = 0; i < ROWS_PER_BLK; ++i) {
        int r = r0 + i;
        if (r >= NN) break;
        size_t idx = (size_t)r * DD + d;
        float v = (hpre[idx] - mu) * rs * g + be;
        hout[idx] = v > 0.f ? v : expm1f(v);
    }
}

// ---------------------------------------------------------------------------
// Pooling + readout
// ---------------------------------------------------------------------------
__global__ void count_kernel(const int* __restrict__ batch, float* __restrict__ counts) {
    int n = blockIdx.x * blockDim.x + threadIdx.x;
    if (n < NN) atomicAdd(&counts[batch[n]], 1.0f);
}
__global__ void __launch_bounds__(256)
pool_kernel(const float* __restrict__ h, const int* __restrict__ batch,
            float* __restrict__ pooled) {
    int n = blockIdx.x;
    int t = threadIdx.x;
    atomicAdd(&pooled[(size_t)batch[n] * DD + t], h[(size_t)n * DD + t]);
}
__global__ void __launch_bounds__(256)
readout_kernel(const float* __restrict__ pooled, const float* __restrict__ counts,
               const float* __restrict__ Wr, const float* __restrict__ br,
               float* __restrict__ out) {
    int t = threadIdx.x;             // 256 = 128 graphs x 2 outputs
    int g = t >> 1, j = t & 1;
    float c = counts[g];
    float inv = 1.0f / (c > 1.0f ? c : 1.0f);
    float acc = 0.f;
    for (int d = 0; d < DD; ++d)
        acc += pooled[(size_t)g * DD + d] * Wr[d * 2 + j];
    out[t] = acc * inv + br[j];
}

// ---------------------------------------------------------------------------
extern "C" void kernel_launch(void* const* d_in, const int* in_sizes, int n_in,
                              void* d_out, int out_size, void* d_ws, size_t ws_size,
                              hipStream_t stream) {
    const float* x     = (const float*)d_in[0];
    const int*   ei    = (const int*)d_in[1];
    const int*   batch = (const int*)d_in[2];
    const float* W[3]  = {(const float*)d_in[3], (const float*)d_in[7], (const float*)d_in[11]};
    const float* bb[3] = {(const float*)d_in[4], (const float*)d_in[8], (const float*)d_in[12]};
    const float* gm[3] = {(const float*)d_in[5], (const float*)d_in[9], (const float*)d_in[13]};
    const float* bt[3] = {(const float*)d_in[6], (const float*)d_in[10], (const float*)d_in[14]};
    const float* Wr    = (const float*)d_in[15];
    const float* br    = (const float*)d_in[16];
    float* out = (float*)d_out;

    char* ws = (char*)d_ws;
    size_t off = 0;
    auto alloc = [&](size_t bytes) -> void* {
        void* p = ws + off;
        off = (off + bytes + 255) & ~(size_t)255;
        return p;
    };

    const size_t NB = (size_t)NN * DD * sizeof(float);   // 51.2 MB
    float*  hbuf[3];
    hbuf[0]        = (float*)alloc(NB);
    hbuf[1]        = (float*)alloc(NB);
    hbuf[2]        = (float*)alloc(NB);
    float*  hw     = (float*)alloc(NB);
    float*  agg    = (float*)alloc(NB);
    float*  deg    = (float*)alloc((size_t)NN * sizeof(float));
    bf16_t* Bh     = (bf16_t*)alloc(65536 * sizeof(bf16_t));
    bf16_t* Bl     = (bf16_t*)alloc(65536 * sizeof(bf16_t));
    float*  stats  = (float*)alloc(512 * sizeof(float));  // sums | sumsq
    float*  pooled = (float*)alloc((size_t)GG * DD * sizeof(float));
    float*  counts = (float*)alloc(GG * sizeof(float));
    (void)ws_size; (void)n_in; (void)in_sizes; (void)out_size;

    // --- degree / normalization ---
    deg_init_kernel<<<(NN + 255) / 256, 256, 0, stream>>>(deg);
    deg_accum_kernel<<<(EE + 255) / 256, 256, 0, stream>>>(ei, deg);
    dinv_kernel<<<(NN + 255) / 256, 256, 0, stream>>>(deg);   // deg now holds dinv

    const int stat_blocks = (NN + ROWS_PER_BLK - 1) / ROWS_PER_BLK;
    const int gemm_blocks = (MTILES + 3) / 4;                 // 4 waves / block

    const float* h    = x;
    const float* prev = nullptr;
    for (int l = 0; l < 3; ++l) {
        pack_w_kernel<<<65536 / 256, 256, 0, stream>>>(W[l], Bh, Bl);
        gemm_bf16x3_kernel<<<gemm_blocks, 128, 0, stream>>>(h, Bh, Bl, hw, MTILES);

        hipMemsetAsync(agg, 0, NB, stream);
        scatter_kernel<<<EE + NN, 256, 0, stream>>>(hw, ei, deg, agg);

        hipMemsetAsync(stats, 0, 512 * sizeof(float), stream);
        bn_stats_kernel<<<stat_blocks, 256, 0, stream>>>(agg, prev, bb[l],
                                                         stats, stats + 256);
        float* hout = hbuf[l];
        bn_apply_kernel<<<stat_blocks, 256, 0, stream>>>(agg, stats, stats + 256,
                                                         gm[l], bt[l], hout);
        prev = h;       // h[:, :D] == h since FEAT == D
        h = hout;
    }

    // --- pooling + readout ---
    hipMemsetAsync(pooled, 0, (size_t)GG * DD * sizeof(float), stream);
    hipMemsetAsync(counts, 0, GG * sizeof(float), stream);
    count_kernel<<<(NN + 255) / 256, 256, 0, stream>>>(batch, counts);
    pool_kernel<<<NN, 256, 0, stream>>>(h, batch, pooled);
    readout_kernel<<<1, 256, 0, stream>>>(pooled, counts, Wr, br, out);
}
